// ConditionalMoELayer_48421461295381
// MI455X (gfx1250) — matrix operand order repaired and checked by reading, not compiled, not run on hardware
//
#include <hip/hip_runtime.h>
#include <stdint.h>

typedef __bf16 bf16x16 __attribute__((ext_vector_type(16)));
typedef __bf16 bf16x8  __attribute__((ext_vector_type(8)));
typedef __bf16 bf16x4  __attribute__((ext_vector_type(4)));
typedef float  f32x8   __attribute__((ext_vector_type(8)));

__device__ __forceinline__ __bf16 f2bf(float f) {
  uint32_t u = __builtin_bit_cast(uint32_t, f);
  uint32_t r = u + 0x7FFFu + ((u >> 16) & 1u);   // round-to-nearest-even
  return __builtin_bit_cast(__bf16, (uint16_t)(r >> 16));
}
__device__ __forceinline__ float bf2f(__bf16 b) {
  uint32_t u = ((uint32_t)__builtin_bit_cast(uint16_t, b)) << 16;
  return __builtin_bit_cast(float, u);
}

#define BM 128
#define BN 128
#define BK 32
#define LDS_STR 40   // bf16 elements per LDS row (32 + 8 pad; 80B rows keep 16B alignment)

// ---------------- elementwise fp32 -> bf16 ----------------
__global__ __launch_bounds__(256) void cvt_kernel(const float* __restrict__ in,
                                                  __bf16* __restrict__ out,
                                                  long n) {
  long i = ((long)blockIdx.x * 256 + threadIdx.x) * 4;
  if (i >= n) return;
  float4 v = *(const float4*)(in + i);
  bf16x4 o;
  o[0] = f2bf(v.x); o[1] = f2bf(v.y); o[2] = f2bf(v.z); o[3] = f2bf(v.w);
  *(bf16x4*)(out + i) = o;
}

// ------------- fp32 [K,N] -> bf16 [N,K] (transpose), batched over z -------------
__global__ __launch_bounds__(256) void cvtT_kernel(const float* __restrict__ in,
                                                   __bf16* __restrict__ out,
                                                   int K, int N) {
  __shared__ __bf16 tile[32][33];
  const long boff = (long)blockIdx.z * K * N;
  in += boff;
  out += boff;
  int k0 = blockIdx.y * 32, n0 = blockIdx.x * 32;
  int tx = threadIdx.x & 31, ty = threadIdx.x >> 5;  // ty: 0..7
#pragma unroll
  for (int r = ty; r < 32; r += 8)
    tile[tx][r] = f2bf(in[(long)(k0 + r) * N + (n0 + tx)]);  // tile[n_loc][k_loc]
  __syncthreads();
#pragma unroll
  for (int r = ty; r < 32; r += 8)
    out[(long)(n0 + r) * K + (k0 + tx)] = tile[r][tx];
}

// ---------------- bf16 GEMM: C[M,N] = f(A[M,K] @ Bt[N,K]^T + bias) ----------------
// Tiles staged memory->LDS with GLOBAL_LOAD_ASYNC_TO_LDS_B128 (ASYNCcnt), double-buffered.
// EPI==0: store bf16 (optional ReLU).  EPI==1: out (+)= scales[row,scale_col]*(acc+bias), fp32.
template <int EPI, bool RELU>
__global__ __launch_bounds__(256) void moe_gemm_kernel(
    const __bf16* __restrict__ A, const __bf16* __restrict__ Bt,
    const float* __restrict__ bias, void* __restrict__ C,
    const float* __restrict__ scales, int scale_col,
    int M, int N, int K, int lda, int ldb, int ldc, int init) {
  __shared__ __align__(16) __bf16 As[2][BM][LDS_STR];
  __shared__ __align__(16) __bf16 Bs[2][BN][LDS_STR];

  const int tid  = threadIdx.x;
  const int lane = tid & 31;
  const int wave = tid >> 5;
  const int wm0 = (wave & 1) * 64;
  const int wn0 = (wave >> 1) * 32;
  const int m0 = blockIdx.y * BM;
  const int n0 = blockIdx.x * BN;
  const int half = lane >> 4;
  const int lrow = lane & 15;

  // staging assignment: each thread owns 4 chunks of 8 bf16 (16B) per K-step
  const int arow = tid >> 2;          // 0..63
  const int acol = (tid & 3) << 3;    // 0,8,16,24

  // per-lane LDS byte addresses (flat shared aperture keeps LDS offset in addr[31:0])
  const uint32_t lA0 = (uint32_t)(uintptr_t)&As[0][arow][acol];
  const uint32_t lA1 = lA0 + (uint32_t)(64 * LDS_STR * 2);
  const uint32_t lB0 = (uint32_t)(uintptr_t)&Bs[0][arow][acol];
  const uint32_t lB1 = lB0 + (uint32_t)(64 * LDS_STR * 2);
  const uint32_t BUFSTEP = (uint32_t)(BM * LDS_STR * 2);   // 10240 B per buffer

  // 32-bit global byte offsets (GVS mode: saddr = matrix base)
  const uint32_t oA0 = (uint32_t)(((long)(m0 + arow) * lda + acol) * 2);
  const uint32_t oA1 = oA0 + (uint32_t)((long)64 * lda * 2);
  const uint32_t oB0 = (uint32_t)(((long)(n0 + arow) * ldb + acol) * 2);
  const uint32_t oB1 = oB0 + (uint32_t)((long)64 * ldb * 2);

  f32x8 acc[4][2];
#pragma unroll
  for (int mf = 0; mf < 4; mf++)
#pragma unroll
    for (int nf = 0; nf < 2; nf++) acc[mf][nf] = (f32x8)0.0f;

  auto issue = [&](int buf, int kstep) {
    const uint32_t ko = (uint32_t)kstep * (uint32_t)(BK * 2);
    const uint32_t bs = buf ? BUFSTEP : 0u;
    asm volatile("global_load_async_to_lds_b128 %0, %1, %2"
                 :: "v"(lA0 + bs), "v"(oA0 + ko), "s"(A) : "memory");
    asm volatile("global_load_async_to_lds_b128 %0, %1, %2"
                 :: "v"(lA1 + bs), "v"(oA1 + ko), "s"(A) : "memory");
    asm volatile("global_load_async_to_lds_b128 %0, %1, %2"
                 :: "v"(lB0 + bs), "v"(oB0 + ko), "s"(Bt) : "memory");
    asm volatile("global_load_async_to_lds_b128 %0, %1, %2"
                 :: "v"(lB1 + bs), "v"(oB1 + ko), "s"(Bt) : "memory");
  };
  auto wsync = [&]() {
    asm volatile("s_wait_asynccnt 0x0" ::: "memory");
    __syncthreads();
  };
  auto compute = [&](int buf) {
    bf16x16 afrag[4];
#pragma unroll
    for (int mf = 0; mf < 4; mf++) {
      const __bf16* pr = &As[buf][wm0 + mf * 16 + lrow][0];
      bf16x8 lo = *(const bf16x8*)(pr + half * 8);        // K = half*8 .. +7
      bf16x8 hi = *(const bf16x8*)(pr + 16 + half * 8);   // K = 16+half*8 .. +7
      afrag[mf] = __builtin_shufflevector(lo, hi, 0, 1, 2, 3, 4, 5, 6, 7, 8, 9,
                                          10, 11, 12, 13, 14, 15);
    }
    bf16x16 bfrag[2];
#pragma unroll
    for (int nf = 0; nf < 2; nf++) {
      const __bf16* pr = &Bs[buf][wn0 + nf * 16 + lrow][0] + half * 8; // K = half*8..+7, 16+half*8..+7
      bf16x8 lo = *(const bf16x8*)pr;
      bf16x8 hi = *(const bf16x8*)(pr + 16);
      bfrag[nf] = __builtin_shufflevector(lo, hi, 0, 1, 2, 3, 4, 5, 6, 7, 8, 9,
                                          10, 11, 12, 13, 14, 15);
    }
#pragma unroll
    for (int mf = 0; mf < 4; mf++)
#pragma unroll
      for (int nf = 0; nf < 2; nf++)
        acc[mf][nf] = __builtin_amdgcn_wmma_f32_16x16x32_bf16(
            false, afrag[mf], false, bfrag[nf], (short)0, acc[mf][nf], false,
            false);
  };

  const int NK = K / BK;   // even for all shapes used here (K = 1024 or 2048)

  issue(0, 0);
  wsync();
  for (int kt = 0; kt < NK; kt += 2) {
    issue(1, kt + 1);      // kt+1 < NK always (NK even)
    compute(0);
    wsync();
    if (kt + 2 < NK) issue(0, kt + 2);
    compute(1);
    if (kt + 2 < NK) wsync();
  }

  // epilogue: C fragment layout: VGPR i -> M = i + 8*half, N = lane%16
#pragma unroll
  for (int mf = 0; mf < 4; mf++) {
#pragma unroll
    for (int nf = 0; nf < 2; nf++) {
      int col = n0 + wn0 + nf * 16 + lrow;
      float bv = bias[col];
#pragma unroll
      for (int i = 0; i < 8; i++) {
        int row = m0 + wm0 + mf * 16 + half * 8 + i;
        float v = acc[mf][nf][i] + bv;
        if constexpr (EPI == 0) {
          if constexpr (RELU) v = fmaxf(v, 0.0f);
          ((__bf16*)C)[(long)row * ldc + col] = f2bf(v);
        } else {
          float w = scales[(long)row * 4 + scale_col];
          float t = w * v;
          float* po = (float*)C + (long)row * ldc + col;
          if (init) *po = t;
          else *po = *po + t;
        }
      }
    }
  }
}

// One wave per token: gate logits, entropy->k, top-k softmax -> dense weights [T,4]
__global__ __launch_bounds__(256) void moe_router_kernel(
    const float* __restrict__ x, const float* __restrict__ gate_w,
    const float* __restrict__ gate_b, const __bf16* __restrict__ hb,
    const float* __restrict__ dp_w2, const float* __restrict__ dp_b2,
    float* __restrict__ dense_w, int T, int D, int DH) {
  const int wave = threadIdx.x >> 5;
  const int lane = threadIdx.x & 31;
  const int t = blockIdx.x * 8 + wave;
  if (t >= T) return;

  const float* xt = x + (long)t * D;
  float l0 = 0.f, l1 = 0.f, l2 = 0.f, l3 = 0.f;
  for (int d = lane; d < D; d += 32) {
    float xv = xt[d];
    float4 g = *(const float4*)(gate_w + (long)d * 4);
    l0 += xv * g.x; l1 += xv * g.y; l2 += xv * g.z; l3 += xv * g.w;
  }
  const __bf16* ht = hb + (long)t * DH;
  float es = 0.f;
  for (int i = lane; i < DH; i += 32) es += bf2f(ht[i]) * dp_w2[i];
#pragma unroll
  for (int off = 16; off > 0; off >>= 1) {
    l0 += __shfl_xor(l0, off, 32);
    l1 += __shfl_xor(l1, off, 32);
    l2 += __shfl_xor(l2, off, 32);
    l3 += __shfl_xor(l3, off, 32);
    es += __shfl_xor(es, off, 32);
  }
  if (lane == 0) {
    float z = es + dp_b2[0];
    float ent = (z > 30.f) ? z : log1pf(expf(z));            // softplus
    float nrm = fminf(fmaxf((ent - 0.5f) / 1.5f, 0.f), 1.f);
    int k = (int)rintf(1.0f + nrm * 3.0f);                   // half-to-even
    k = k < 1 ? 1 : (k > 4 ? 4 : k);

    float lg[4] = {l0 + gate_b[0], l1 + gate_b[1], l2 + gate_b[2],
                   l3 + gate_b[3]};
    int idx[4] = {0, 1, 2, 3};
#pragma unroll
    for (int a = 0; a < 3; a++) {
      int best = a;
#pragma unroll
      for (int b2 = 0; b2 < 4; b2++)
        if (b2 > a && lg[idx[b2]] > lg[idx[best]]) best = b2;  // strict >: stable ties
      int tmp = idx[a]; idx[a] = idx[best]; idx[best] = tmp;
    }
    float mx = lg[idx[0]];
    float w[4] = {0.f, 0.f, 0.f, 0.f};
    float se = 0.f;
    for (int j = 0; j < 4; j++)
      if (j < k) { w[j] = expf(lg[idx[j]] - mx); se += w[j]; }
    float o[4] = {0.f, 0.f, 0.f, 0.f};
    for (int j = 0; j < 4; j++)
      if (j < k) o[idx[j]] = w[j] / se;
    float* dw = dense_w + (long)t * 4;
    dw[0] = o[0]; dw[1] = o[1]; dw[2] = o[2]; dw[3] = o[3];
  }
}

extern "C" void kernel_launch(void* const* d_in, const int* in_sizes, int n_in,
                              void* d_out, int out_size, void* d_ws,
                              size_t ws_size, hipStream_t stream) {
  const float* x      = (const float*)d_in[0];
  const float* gate_w = (const float*)d_in[1];
  const float* gate_b = (const float*)d_in[2];
  const float* dp_w1  = (const float*)d_in[3];
  const float* dp_b1  = (const float*)d_in[4];
  const float* dp_w2  = (const float*)d_in[5];
  const float* dp_b2  = (const float*)d_in[6];
  const float* ew1    = (const float*)d_in[7];
  const float* eb1    = (const float*)d_in[8];
  const float* ew2    = (const float*)d_in[9];
  const float* eb2    = (const float*)d_in[10];

  const int D = 1024, E = 4, H = 2048, DH = 512;
  const int T = in_sizes[0] / D;  // 8192

  // workspace layout (bytes)
  char* ws = (char*)d_ws;
  size_t off = 0;
  __bf16* hidb  = (__bf16*)(ws + off); off += (size_t)T * H * 2;       // [T,H]
  __bf16* hb    = (__bf16*)(ws + off); off += (size_t)T * DH * 2;      // [T,DH]
  float*  dw    = (float*) (ws + off); off += (size_t)T * E * 4;       // [T,E]
  __bf16* xb    = (__bf16*)(ws + off); off += (size_t)T * D * 2;       // [T,D]
  __bf16* w1T   = (__bf16*)(ws + off); off += (size_t)DH * D * 2;      // [DH,D]
  __bf16* e1T   = (__bf16*)(ws + off); off += (size_t)E * H * D * 2;   // [E][H,D]
  __bf16* e2T   = (__bf16*)(ws + off); off += (size_t)E * D * H * 2;   // [E][D,H]
  float*  out   = (float*)d_out;                                       // [T,D]

  dim3 blk(256);

  // one-time conversions / transposes (bf16, weights N-major)
  {
    long nx = (long)T * D;
    cvt_kernel<<<dim3((unsigned)((nx / 4 + 255) / 256)), blk, 0, stream>>>(x, xb, nx);
    cvtT_kernel<<<dim3(DH / 32, D / 32, 1), blk, 0, stream>>>(dp_w1, w1T, D, DH);
    cvtT_kernel<<<dim3(H / 32, D / 32, E), blk, 0, stream>>>(ew1, e1T, D, H);
    cvtT_kernel<<<dim3(D / 32, H / 32, E), blk, 0, stream>>>(ew2, e2T, H, D);
  }

  // difficulty hidden layer: h = relu(x @ dp_w1 + dp_b1) -> bf16
  moe_gemm_kernel<0, true><<<dim3(DH / BN, T / BM), blk, 0, stream>>>(
      xb, w1T, dp_b1, hb, nullptr, 0, T, DH, D, D, D, DH, 0);

  // router: entropy -> k, top-k softmax -> dense weights
  moe_router_kernel<<<dim3(T / 8), blk, 0, stream>>>(x, gate_w, gate_b, hb,
                                                     dp_w2, dp_b2, dw, T, D, DH);

  for (int e = 0; e < E; e++) {
    // hid = relu(x @ ew1[e] + eb1[e]) -> bf16
    moe_gemm_kernel<0, true><<<dim3(H / BN, T / BM), blk, 0, stream>>>(
        xb, e1T + (size_t)e * H * D, eb1 + (size_t)e * H, hidb, nullptr, 0, T,
        H, D, D, D, H, 0);
    // out (+)= dense_w[:,e] * (hid @ ew2[e] + eb2[e])
    moe_gemm_kernel<1, false><<<dim3(D / BN, T / BM), blk, 0, stream>>>(
        hidb, e2T + (size_t)e * D * H, eb2 + (size_t)e * D, out, dw, e, T, D, H,
        H, H, D, (e == 0) ? 1 : 0);
  }
}
